// IDST_1133871366831
// MI455X (gfx1250) — matrix operand order repaired and checked by reading, not compiled, not run on hardware
//
#include <hip/hip_runtime.h>

// ---------------------------------------------------------------------------
// IDST (DST-III) of a 4096x4096 fp32 matrix along rows, as Y = X * M with
// M[n][k] = w_n * sin(pi*(n+1)*(2k+1)/(2N)),  w_n = 2 (n<N-1), 1 (n==N-1).
// Computed as a split-bf16 (hi+lo) WMMA GEMM with fp32 accumulation:
//   Y = Xhi*Mhi + Xhi*Mlo + Xlo*Mhi   (lo*lo term negligible)
// M is staged TRANSPOSED (Mt[col][n]) so both A and B tiles are contiguous-K
// 16-byte copies into LDS. All staged operands (128 MB) fit in MI455X's
// 192 MB L2, so the GEMM runs WMMA-throughput-bound, not HBM-bound.
// ---------------------------------------------------------------------------

typedef __bf16 bf16_t;
typedef __attribute__((ext_vector_type(8)))  __bf16 bf16x8;   // 16 bytes
typedef __attribute__((ext_vector_type(16))) __bf16 v16bf;    // 32 bytes (8 VGPRs)
typedef __attribute__((ext_vector_type(8)))  float  v8f;      // 32 bytes (8 VGPRs)

#define NDIM 4096
#define BM 128
#define BN 128
#define BK 32
#define NITER (NDIM / BK)

// --------------------------- split fp32 -> bf16 hi/lo ----------------------
__global__ void split_bf16_kernel(const float* __restrict__ x,
                                  bf16_t* __restrict__ hi,
                                  bf16_t* __restrict__ lo,
                                  int use_lo) {
    size_t idx = (size_t)blockIdx.x * blockDim.x + threadIdx.x;
    float v = x[idx];
    bf16_t h = (bf16_t)v;
    hi[idx] = h;
    if (use_lo) lo[idx] = (bf16_t)(v - (float)h);
}

// ------------------- generate DST-III matrix, TRANSPOSED -------------------
// Mt[col][n] = w_n * sin(pi*(n+1)*(2*col+1)/(2N)).
// Exact integer phase: sin(pi*m/(2N)) has period 4N in m; 4N = 16384 = 2^14.
__global__ void gen_dst3t_kernel(bf16_t* __restrict__ hi,
                                 bf16_t* __restrict__ lo,
                                 int use_lo) {
    size_t idx = (size_t)blockIdx.x * blockDim.x + threadIdx.x;
    int col = (int)(idx >> 12);      // 0..4095  (output column of the DST)
    int n   = (int)(idx & 4095);     // 0..4095  (K dimension of GEMM)
    unsigned m = ((unsigned)(n + 1) * (unsigned)(2 * col + 1)) & 16383u;
    float ang = (float)m * 1.9174759848570515e-4f;   // pi / 16384
    float v = sinf(ang);
    if (n != NDIM - 1) v *= 2.0f;
    bf16_t h = (bf16_t)v;
    hi[idx] = h;
    if (use_lo) lo[idx] = (bf16_t)(v - (float)h);
}

// --------------------------- WMMA GEMM kernel ------------------------------
// Y[4096x4096] = X * M, fp32 out, bf16x2 inputs, M given transposed.
// Block: 256 threads (8 wave32s), tile 128x128, K-step 32, LDS double-buffer.
// Wave grid 4x2: each wave -> 32 rows x 64 cols = 2x4 fragments of 16x16.
template <bool USE_LO>
__global__ __launch_bounds__(256) void dst_wmma_gemm(
    const bf16_t* __restrict__ Ahi, const bf16_t* __restrict__ Alo,
    const bf16_t* __restrict__ Bhi, const bf16_t* __restrict__ Blo,
    float* __restrict__ out) {

    __shared__ bf16_t sAhi[2][BM][BK];                 // 16 KB
    __shared__ bf16_t sBhi[2][BN][BK];                 // 16 KB ([col][k])
    __shared__ bf16_t sAlo[USE_LO ? 2 : 1][BM][BK];
    __shared__ bf16_t sBlo[USE_LO ? 2 : 1][BN][BK];

    const int tid  = threadIdx.x;
    const int wave = tid >> 5;
    const int lane = tid & 31;
    const int wm   = wave >> 1;            // 0..3
    const int wn   = wave & 1;             // 0..1
    const int rowBase = blockIdx.y * BM;
    const int colBase = blockIdx.x * BN;

    const int r     = lane & 15;           // row/col within 16x16 tile
    const int hi16  = lane >> 4;           // lane half
    const int khalf = hi16 * 8;            // unused: A and B frags both load from kbase
    const int kbase = hi16 * 16;           // B frag: K offset (16 contiguous)

    // Per-thread staging-copy coordinates (2 chunks of 8 bf16 for A and B).
    int ar[2], ac[2], br[2], bc[2];
    size_t aOff[2], bOff[2];
#pragma unroll
    for (int i = 0; i < 2; ++i) {
        int chunk = tid + i * 256;                 // 0..511
        ar[i] = chunk >> 2;                        // 0..127 (row of X)
        ac[i] = (chunk & 3) * 8;                   // 0,8,16,24 (k)
        br[i] = chunk >> 2;                        // 0..127 (col of Mt)
        bc[i] = (chunk & 3) * 8;                   // 0,8,16,24 (k)
        aOff[i] = (size_t)(rowBase + ar[i]) * NDIM + ac[i];
        bOff[i] = (size_t)(colBase + br[i]) * NDIM + bc[i];
    }

    v8f acc[2][4];
    const v8f zf = {0.f, 0.f, 0.f, 0.f, 0.f, 0.f, 0.f, 0.f};
#pragma unroll
    for (int i = 0; i < 2; ++i)
#pragma unroll
        for (int j = 0; j < 4; ++j) acc[i][j] = zf;

    bf16x8 rAh[2], rBh[2], rAl[2], rBl[2];

    // ---- preload tile 0 into LDS buffer 0 --------------------------------
#pragma unroll
    for (int i = 0; i < 2; ++i) {
        rAh[i] = *(const bf16x8*)(Ahi + aOff[i]);
        rBh[i] = *(const bf16x8*)(Bhi + bOff[i]);
        if constexpr (USE_LO) {
            rAl[i] = *(const bf16x8*)(Alo + aOff[i]);
            rBl[i] = *(const bf16x8*)(Blo + bOff[i]);
        }
    }
#pragma unroll
    for (int i = 0; i < 2; ++i) {
        *(bf16x8*)&sAhi[0][ar[i]][ac[i]] = rAh[i];
        *(bf16x8*)&sBhi[0][br[i]][bc[i]] = rBh[i];
        if constexpr (USE_LO) {
            *(bf16x8*)&sAlo[0][ar[i]][ac[i]] = rAl[i];
            *(bf16x8*)&sBlo[0][br[i]][bc[i]] = rBl[i];
        }
    }
    __syncthreads();

    for (int it = 0; it < NITER; ++it) {
        const int cur = it & 1;
        const int nxt = cur ^ 1;
        const bool more = (it + 1) < NITER;

        // ---- issue next tile's global loads (latency hidden by WMMAs) ----
        if (more) {
            size_t k1 = (size_t)(it + 1) * BK;
#pragma unroll
            for (int i = 0; i < 2; ++i) {
                rAh[i] = *(const bf16x8*)(Ahi + aOff[i] + k1);
                rBh[i] = *(const bf16x8*)(Bhi + bOff[i] + k1);
                if constexpr (USE_LO) {
                    rAl[i] = *(const bf16x8*)(Alo + aOff[i] + k1);
                    rBl[i] = *(const bf16x8*)(Blo + bOff[i] + k1);
                }
            }
        }

        // ---- load fragments (A and B both contiguous k = 16*hi16 + s, wave32) -------------
        v16bf afh[2], afl[2], bfh[4], bfl[4];
#pragma unroll
        for (int tm = 0; tm < 2; ++tm) {
            int arow = wm * 32 + tm * 16 + r;
            ((bf16x8*)&afh[tm])[0] = *(const bf16x8*)&sAhi[cur][arow][kbase];
            ((bf16x8*)&afh[tm])[1] = *(const bf16x8*)&sAhi[cur][arow][kbase + 8];
            if constexpr (USE_LO) {
                ((bf16x8*)&afl[tm])[0] = *(const bf16x8*)&sAlo[cur][arow][kbase];
                ((bf16x8*)&afl[tm])[1] = *(const bf16x8*)&sAlo[cur][arow][kbase + 8];
            }
        }
#pragma unroll
        for (int tn = 0; tn < 4; ++tn) {
            int bcol = wn * 64 + tn * 16 + r;
            bfh[tn] = *(const v16bf*)&sBhi[cur][bcol][kbase];
            if constexpr (USE_LO) bfl[tn] = *(const v16bf*)&sBlo[cur][bcol][kbase];
        }

        // ---- 16x16x32 bf16 WMMA, fp32 accumulate -------------------------
#pragma unroll
        for (int tm = 0; tm < 2; ++tm)
#pragma unroll
            for (int tn = 0; tn < 4; ++tn) {
                acc[tm][tn] = __builtin_amdgcn_wmma_f32_16x16x32_bf16(
                    false, afh[tm], false, bfh[tn], (short)0, acc[tm][tn],
                    false, false);
                if constexpr (USE_LO) {
                    acc[tm][tn] = __builtin_amdgcn_wmma_f32_16x16x32_bf16(
                        false, afh[tm], false, bfl[tn], (short)0, acc[tm][tn],
                        false, false);
                    acc[tm][tn] = __builtin_amdgcn_wmma_f32_16x16x32_bf16(
                        false, afl[tm], false, bfh[tn], (short)0, acc[tm][tn],
                        false, false);
                }
            }

        // ---- commit next tile to the alternate LDS buffer ----------------
        if (more) {
#pragma unroll
            for (int i = 0; i < 2; ++i) {
                *(bf16x8*)&sAhi[nxt][ar[i]][ac[i]] = rAh[i];
                *(bf16x8*)&sBhi[nxt][br[i]][bc[i]] = rBh[i];
                if constexpr (USE_LO) {
                    *(bf16x8*)&sAlo[nxt][ar[i]][ac[i]] = rAl[i];
                    *(bf16x8*)&sBlo[nxt][br[i]][bc[i]] = rBl[i];
                }
            }
        }
        __syncthreads();
    }

    // ---- write back per C/D f32 layout: VGPR j -> M = j + 8*hi16 ---------
#pragma unroll
    for (int tm = 0; tm < 2; ++tm)
#pragma unroll
        for (int tn = 0; tn < 4; ++tn) {
            int row0 = rowBase + wm * 32 + tm * 16;
            int col  = colBase + wn * 64 + tn * 16 + r;
#pragma unroll
            for (int j = 0; j < 8; ++j) {
                int m = j + 8 * hi16;
                out[(size_t)(row0 + m) * NDIM + col] = acc[tm][tn][j];
            }
        }
}

// --------------------------- naive fallback (no ws) ------------------------
__global__ void idst_direct_kernel(const float* __restrict__ x,
                                   float* __restrict__ y) {
    int rrow = blockIdx.y;
    int k = blockIdx.x * blockDim.x + threadIdx.x;
    const float* xr = x + (size_t)rrow * NDIM;
    float s = 0.f;
    for (int n = 0; n < NDIM; ++n) {
        unsigned m = ((unsigned)(n + 1) * (unsigned)(2 * k + 1)) & 16383u;
        float w = (n == NDIM - 1) ? 1.f : 2.f;
        s += xr[n] * w * __sinf((float)m * 1.9174759848570515e-4f);
    }
    y[(size_t)rrow * NDIM + k] = s;
}

// ---------------------------------------------------------------------------
extern "C" void kernel_launch(void* const* d_in, const int* in_sizes, int n_in,
                              void* d_out, int out_size, void* d_ws, size_t ws_size,
                              hipStream_t stream) {
    const float* x = (const float*)d_in[0];
    float* out = (float*)d_out;

    const size_t nelem = (size_t)NDIM * NDIM;
    const size_t bufB  = nelem * sizeof(bf16_t);   // 32 MB per buffer

    if (ws_size < 2 * bufB) {
        // No room to stage bf16 operands: exact (slow) direct evaluation.
        dim3 g(NDIM / 256, NDIM);
        idst_direct_kernel<<<g, 256, 0, stream>>>(x, out);
        return;
    }

    const int use_lo = (ws_size >= 4 * bufB) ? 1 : 0;
    char* w = (char*)d_ws;
    bf16_t* Xhi = (bf16_t*)(w);
    bf16_t* Mhi = (bf16_t*)(w + bufB);
    bf16_t* Xlo = use_lo ? (bf16_t*)(w + 2 * bufB) : Xhi;
    bf16_t* Mlo = use_lo ? (bf16_t*)(w + 3 * bufB) : Mhi;

    const int threads = 256;
    const int blocks  = (int)(nelem / threads);
    split_bf16_kernel<<<blocks, threads, 0, stream>>>(x, Xhi, Xlo, use_lo);
    gen_dst3t_kernel<<<blocks, threads, 0, stream>>>(Mhi, Mlo, use_lo);

    dim3 grid(NDIM / BN, NDIM / BM);   // 32 x 32 workgroups
    if (use_lo)
        dst_wmma_gemm<true><<<grid, 256, 0, stream>>>(Xhi, Xlo, Mhi, Mlo, out);
    else
        dst_wmma_gemm<false><<<grid, 256, 0, stream>>>(Xhi, Xlo, Mhi, Mlo, out);
}
